// HyperSharedBlock_87170656239974
// MI455X (gfx1250) — matrix operand order repaired and checked by reading, not compiled, not run on hardware
//
#include <hip/hip_runtime.h>
#include <hip/hip_bf16.h>
#include <cstdint>
#include <cstddef>

typedef __attribute__((ext_vector_type(16))) __bf16 v16bf;
typedef __attribute__((ext_vector_type(8)))  __bf16 v8bf;
typedef __attribute__((ext_vector_type(8)))  float  v8f;

#define DEV __device__ __forceinline__

constexpr int Bb   = 2;
constexpr int Ss   = 2048;
constexpr int DIMc = 1024;
constexpr int NH   = 16;
constexpr int NKV  = 4;
constexpr int HD   = 64;
constexpr int FFNc = 4096;
constexpr int ROWS = Bb * Ss;   // 4096 token rows

// ---------- bf16 helpers (RNE convert, bit-exact storage) ----------
DEV __bf16 f2bf(float f) {
  unsigned u = __builtin_bit_cast(unsigned, f);
  unsigned r = (u + 0x7FFFu + ((u >> 16) & 1u)) >> 16;
  unsigned short s = (unsigned short)r;
  return __builtin_bit_cast(__bf16, s);
}

// ---------- WMMA wrapper ----------
DEV v8f wmma_bf16(v16bf a, v16bf b, v8f c) {
  return __builtin_amdgcn_wmma_f32_16x16x32_bf16(
      /*neg_a=*/false, a, /*neg_b=*/false, b,
      /*c_mod=*/(short)0, c, /*reuse_a=*/false, /*reuse_b=*/false);
}

// A fragment: 16x32 bf16, row-major source. Lane L holds row M=L%16.
// K indices: lanes 0-15 -> {k0..k0+7, k0+16..k0+23}; lanes 16-31 -> +8.
DEV v16bf load_frag_a(const __bf16* A, int lda, int row0, int k0) {
  int lane = threadIdx.x & 31;
  const __bf16* p = A + (size_t)(row0 + (lane & 15)) * lda + k0 + ((lane >> 4) << 3);
  v8bf lo = *(const v8bf*)p;
  v8bf hi = *(const v8bf*)(p + 16);
  v16bf o;
#pragma unroll
  for (int i = 0; i < 8; ++i) { o[i] = lo[i]; o[i + 8] = hi[i]; }
  return o;
}

// B fragment: 32x16 bf16, supplied TRANSPOSED (Bt is [N][K] row-major).
// Lane L holds column N=L%16; K indices same as A: {k0..k0+7, k0+16..k0+23}, +8 for L>=16.
DEV v16bf load_frag_b(const __bf16* Bt, int ldb, int col0, int k0) {
  int lane = threadIdx.x & 31;
  const __bf16* p = Bt + (size_t)(col0 + (lane & 15)) * ldb + k0 + ((lane >> 4) << 3);
  v8bf lo = *(const v8bf*)p;
  v8bf hi = *(const v8bf*)(p + 16);
  v16bf o;
#pragma unroll
  for (int i = 0; i < 8; ++i) { o[i] = lo[i]; o[i + 8] = hi[i]; }
  return o;
}

// half-wave (16-lane) reductions: xor masks 1,2,4,8 stay inside each half,
// so lanes 0-15 reduce rows r, lanes 16-31 reduce rows r+8 independently.
DEV float halfmax(float v) {
#pragma unroll
  for (int m = 1; m < 16; m <<= 1) v = fmaxf(v, __shfl_xor(v, m, 32));
  return v;
}
DEV float halfsum(float v) {
#pragma unroll
  for (int m = 1; m < 16; m <<= 1) v += __shfl_xor(v, m, 32);
  return v;
}

// ======================= elementwise kernels =======================

// Wt[n][k] = bf16(W[k][n])   (W is [K][N] fp32 row-major)
__global__ void transpose_w_kernel(const float* __restrict__ W,
                                   __bf16* __restrict__ Wt, int K, int N) {
  size_t idx = (size_t)blockIdx.x * blockDim.x + threadIdx.x;
  if (idx >= (size_t)K * N) return;
  int k = (int)(idx / N), n = (int)(idx % N);
  Wt[(size_t)n * K + k] = f2bf(W[idx]);
}

// row-wise RMSNorm: out_bf16[row] = bf16(x * rsqrt(mean(x^2)+eps) * w)
__global__ void rmsnorm_kernel(const float* __restrict__ x,
                               const float* __restrict__ w,
                               __bf16* __restrict__ out, int D) {
  int row = blockIdx.x;
  const float* xr = x + (size_t)row * D;
  float ss = 0.f;
  for (int i = threadIdx.x; i < D; i += blockDim.x) { float v = xr[i]; ss += v * v; }
#pragma unroll
  for (int m = 1; m < 32; m <<= 1) ss += __shfl_xor(ss, m, 32);
  __shared__ float red[8];
  if ((threadIdx.x & 31) == 0) red[threadIdx.x >> 5] = ss;
  __syncthreads();
  if (threadIdx.x < 8) {
    float v = red[threadIdx.x];
#pragma unroll
    for (int m = 1; m < 8; m <<= 1) v += __shfl_xor(v, m, 32);
    if (threadIdx.x == 0) red[0] = v;
  }
  __syncthreads();
  float rs = rsqrtf(red[0] / (float)D + 1e-6f);
  for (int i = threadIdx.x; i < D; i += blockDim.x)
    out[(size_t)row * D + i] = f2bf(xr[i] * rs * w[i]);
}

// RoPE: src fp32 [B*S, nheads*64] -> dst bf16 [B, nheads, S, 64]
__global__ void rope_kernel(const float* __restrict__ src,
                            const float* __restrict__ cosT,
                            const float* __restrict__ sinT,
                            __bf16* __restrict__ dst, int nheads) {
  int idx = blockIdx.x * blockDim.x + threadIdx.x;   // B*nheads*S*32
  int d = idx & 31;
  int t = idx >> 5;
  int s = t & (Ss - 1);
  t >>= 11;                                          // /Ss
  int h = t % nheads;
  int b = t / nheads;
  const float* row = src + ((size_t)b * Ss + s) * (nheads * HD) + h * HD;
  float x1 = row[d], x2 = row[d + 32];
  float c = cosT[s * 32 + d], sn = sinT[s * 32 + d];
  __bf16* o = dst + ((size_t)(b * nheads + h) * Ss + s) * HD;
  o[d]      = f2bf(x1 * c + x2 * sn);
  o[d + 32] = f2bf(x2 * c - x1 * sn);
}

// V transpose: vf fp32 [B*S, NKV*64] -> vt bf16 [B, NKV, 64, S]
__global__ void vtrans_kernel(const float* __restrict__ vf,
                              __bf16* __restrict__ vt) {
  size_t idx = (size_t)blockIdx.x * blockDim.x + threadIdx.x; // B*NKV*64*S
  int s = (int)(idx & (Ss - 1));
  size_t t = idx >> 11;
  int d  = (int)(t % HD); t /= HD;
  int kv = (int)(t % NKV);
  int b  = (int)(t / NKV);
  vt[idx] = f2bf(vf[((size_t)b * Ss + s) * (NKV * HD) + kv * HD + d]);
}

// ======================= WMMA GEMM kernels =======================
// C[M,N] = A[M,K](bf16 rowmajor) * Bt[N,K](bf16, pre-transposed)
// Block: 256 thr = 8 waves, tile 256(M) x 64(N); each wave 32x64:
// 2 A-frags + 8 accumulators -> 8 WMMA per 12 b128 loads per k-step.
// EPI 0: C = acc (fp32).   EPI 1: C = resid + acc * scale[col].
template <int EPI>
__global__ __launch_bounds__(256) void gemm_bf16_kernel(
    const __bf16* __restrict__ A, const __bf16* __restrict__ Bt,
    float* __restrict__ C, const float* __restrict__ resid,
    const float* __restrict__ scale, int M, int N, int K) {
  int w = threadIdx.x >> 5;
  int row0 = blockIdx.y * 256 + w * 32;
  int col0 = blockIdx.x * 64;
  v8f acc0[4] = {}, acc1[4] = {};
  for (int k0 = 0; k0 < K; k0 += 32) {
    v16bf a0 = load_frag_a(A, K, row0, k0);
    v16bf a1 = load_frag_a(A, K, row0 + 16, k0);
#pragma unroll
    for (int j = 0; j < 4; ++j) {
      v16bf b = load_frag_b(Bt, K, col0 + 16 * j, k0);
      acc0[j] = wmma_bf16(a0, b, acc0[j]);
      acc1[j] = wmma_bf16(a1, b, acc1[j]);
    }
  }
  int lane = threadIdx.x & 31;
  int n = lane & 15;
#pragma unroll
  for (int mi = 0; mi < 2; ++mi) {
    int mb = row0 + mi * 16 + ((lane >> 4) << 3);
#pragma unroll
    for (int j = 0; j < 4; ++j) {
      const v8f& acc = (mi == 0) ? acc0[j] : acc1[j];
#pragma unroll
      for (int r = 0; r < 8; ++r) {
        size_t idx = (size_t)(mb + r) * N + col0 + 16 * j + n;
        if (EPI == 0) C[idx] = acc[r];
        else          C[idx] = resid[idx] + acc[r] * scale[col0 + 16 * j + n];
      }
    }
  }
}

// fused gate/up: H = bf16( silu(A*WgT) * (A*WuT) ); wave tile 32x64,
// A-frags shared by both weight matrices -> 16 WMMA per 20 loads per k-step.
__global__ __launch_bounds__(256) void gemm_gateup_kernel(
    const __bf16* __restrict__ A, const __bf16* __restrict__ WgT,
    const __bf16* __restrict__ WuT, __bf16* __restrict__ H, int N, int K) {
  int w = threadIdx.x >> 5;
  int row0 = blockIdx.y * 256 + w * 32;
  int col0 = blockIdx.x * 64;
  v8f ag0[4] = {}, ag1[4] = {}, au0[4] = {}, au1[4] = {};
  for (int k0 = 0; k0 < K; k0 += 32) {
    v16bf a0 = load_frag_a(A, K, row0, k0);
    v16bf a1 = load_frag_a(A, K, row0 + 16, k0);
#pragma unroll
    for (int j = 0; j < 4; ++j) {
      v16bf bg = load_frag_b(WgT, K, col0 + 16 * j, k0);
      ag0[j] = wmma_bf16(a0, bg, ag0[j]);
      ag1[j] = wmma_bf16(a1, bg, ag1[j]);
      v16bf bu = load_frag_b(WuT, K, col0 + 16 * j, k0);
      au0[j] = wmma_bf16(a0, bu, au0[j]);
      au1[j] = wmma_bf16(a1, bu, au1[j]);
    }
  }
  int lane = threadIdx.x & 31;
  int n = lane & 15;
#pragma unroll
  for (int mi = 0; mi < 2; ++mi) {
    int mb = row0 + mi * 16 + ((lane >> 4) << 3);
#pragma unroll
    for (int j = 0; j < 4; ++j) {
      const v8f& g8 = (mi == 0) ? ag0[j] : ag1[j];
      const v8f& u8 = (mi == 0) ? au0[j] : au1[j];
#pragma unroll
      for (int r = 0; r < 8; ++r) {
        float g = g8[r], u = u8[r];
        float sil = g / (1.f + __expf(-g));
        H[(size_t)(mb + r) * N + col0 + 16 * j + n] = f2bf(sil * u);
      }
    }
  }
}

// ======================= flash attention =======================
// 1 wave = 16 q rows; 32-key tiles; online softmax; causal.
// qbf [B,NH,S,64], kbf [B,NKV,S,64], vt [B,NKV,64,S]; out attn bf16 [B*S, NH*64]
__global__ __launch_bounds__(128) void flash_attn_kernel(
    const __bf16* __restrict__ qbf, const __bf16* __restrict__ kbf,
    const __bf16* __restrict__ vt, __bf16* __restrict__ attn) {
  __shared__ float Pbuf[4][16 * 32];   // per-wave transpose slab
  int w = threadIdx.x >> 5;
  int lane = threadIdx.x & 31;
  int tile = blockIdx.x * 4 + w;       // B*NH*(S/16) tiles
  int qt = tile & 127;                 // S/16 = 128
  int h  = (tile >> 7) & 15;
  int b  = tile >> 11;
  int q0 = qt * 16;

  const __bf16* Q  = qbf + (size_t)(b * NH + h) * Ss * HD;
  const __bf16* Kh = kbf + (size_t)(b * NKV + (h >> 2)) * Ss * HD;
  const __bf16* Vh = vt  + (size_t)(b * NKV + (h >> 2)) * HD * Ss;

  v16bf qa0 = load_frag_a(Q, HD, q0, 0);
  v16bf qa1 = load_frag_a(Q, HD, q0, 32);

  v8f O[4] = {};
  float mrow[8], lrow[8], alpha[8];
#pragma unroll
  for (int r = 0; r < 8; ++r) { mrow[r] = -1e30f; lrow[r] = 0.f; }
  int mb = (lane >> 4) << 3;           // row offset within tile (0 or 8)
  int nc = lane & 15;                  // column within 16-chunk
  float* P = Pbuf[w];

  for (int kt = 0; kt < q0 + 16; kt += 32) {
    // ---- S = Q K^T (two 16x16 column chunks) ----
    v8f s0 = {}, s1 = {};
    {
      v16bf bk = load_frag_b(Kh, HD, kt, 0);
      s0 = wmma_bf16(qa0, bk, s0);
      bk = load_frag_b(Kh, HD, kt, 32);
      s0 = wmma_bf16(qa1, bk, s0);
      bk = load_frag_b(Kh, HD, kt + 16, 0);
      s1 = wmma_bf16(qa0, bk, s1);
      bk = load_frag_b(Kh, HD, kt + 16, 32);
      s1 = wmma_bf16(qa1, bk, s1);
    }
    // ---- scale + causal mask (D layout: lane->col, vgpr r -> row) ----
    const float sc = 0.125f;           // 1/sqrt(64)
#pragma unroll
    for (int r = 0; r < 8; ++r) {
      int row = q0 + mb + r;
      s0[r] = (kt + nc      <= row) ? s0[r] * sc : -1e30f;
      s1[r] = (kt + 16 + nc <= row) ? s1[r] * sc : -1e30f;
    }
    // ---- online softmax ----
#pragma unroll
    for (int r = 0; r < 8; ++r) {
      float rm = halfmax(fmaxf(s0[r], s1[r]));
      float mn = fmaxf(mrow[r], rm);
      alpha[r] = __expf(mrow[r] - mn);
      mrow[r] = mn;
      s0[r] = __expf(s0[r] - mn);
      s1[r] = __expf(s1[r] - mn);
      lrow[r] = lrow[r] * alpha[r] + halfsum(s0[r] + s1[r]);
    }
#pragma unroll
    for (int j = 0; j < 4; ++j)
#pragma unroll
      for (int r = 0; r < 8; ++r) O[j][r] *= alpha[r];
    // ---- transpose P through LDS (same-wave DS is in-order) ----
#pragma unroll
    for (int r = 0; r < 8; ++r) {
      P[(mb + r) * 32 + nc]      = s0[r];
      P[(mb + r) * 32 + 16 + nc] = s1[r];
    }
    v16bf pa;
    {
      int rowA = lane & 15;
      int g = (lane >> 4) << 3;
#pragma unroll
      for (int j = 0; j < 16; ++j) {
        int kk = g + ((j < 8) ? j : j + 8);
        pa[j] = f2bf(P[rowA * 32 + kk]);
      }
    }
    // ---- O += P V ----
#pragma unroll
    for (int j = 0; j < 4; ++j) {
      v16bf bv = load_frag_b(Vh, Ss, 16 * j, kt);
      O[j] = wmma_bf16(pa, bv, O[j]);
    }
  }
  // ---- normalize + store merged [b*S+s, h*64+d] bf16 ----
#pragma unroll
  for (int j = 0; j < 4; ++j) {
#pragma unroll
    for (int r = 0; r < 8; ++r) {
      float o = O[j][r] / lrow[r];
      size_t rowi = (size_t)b * Ss + q0 + mb + r;
      attn[rowi * (NH * HD) + h * HD + 16 * j + nc] = f2bf(o);
    }
  }
}

// ======================= host launch =======================
extern "C" void kernel_launch(void* const* d_in, const int* in_sizes, int n_in,
                              void* d_out, int out_size, void* d_ws, size_t ws_size,
                              hipStream_t stream) {
  (void)in_sizes; (void)n_in; (void)out_size; (void)ws_size;
  const float* x        = (const float*)d_in[0];
  const float* w_norm1  = (const float*)d_in[1];
  const float* wq       = (const float*)d_in[2];
  const float* wk       = (const float*)d_in[3];
  const float* wv       = (const float*)d_in[4];
  const float* wo       = (const float*)d_in[5];
  const float* a_scale  = (const float*)d_in[6];
  const float* w_norm2  = (const float*)d_in[7];
  const float* wg       = (const float*)d_in[8];
  const float* wu       = (const float*)d_in[9];
  const float* wd       = (const float*)d_in[10];
  const float* m_scale  = (const float*)d_in[11];
  const float* cosT     = (const float*)d_in[12];
  const float* sinT     = (const float*)d_in[13];
  float* out = (float*)d_out;

  char* p = (char*)d_ws;
  auto alloc = [&](size_t bytes) -> void* {
    void* r = (void*)p;
    p += (bytes + 255) & ~(size_t)255;
    return r;
  };
  __bf16* wqT   = (__bf16*)alloc((size_t)DIMc * (NH * HD) * 2);
  __bf16* wkT   = (__bf16*)alloc((size_t)DIMc * (NKV * HD) * 2);
  __bf16* wvT   = (__bf16*)alloc((size_t)DIMc * (NKV * HD) * 2);
  __bf16* woT   = (__bf16*)alloc((size_t)(NH * HD) * DIMc * 2);
  __bf16* wgT   = (__bf16*)alloc((size_t)DIMc * FFNc * 2);
  __bf16* wuT   = (__bf16*)alloc((size_t)DIMc * FFNc * 2);
  __bf16* wdT   = (__bf16*)alloc((size_t)FFNc * DIMc * 2);
  __bf16* xnbf  = (__bf16*)alloc((size_t)ROWS * DIMc * 2);
  float*  qf    = (float*) alloc((size_t)ROWS * (NH * HD) * 4);
  float*  kf    = (float*) alloc((size_t)ROWS * (NKV * HD) * 4);
  float*  vf    = (float*) alloc((size_t)ROWS * (NKV * HD) * 4);
  __bf16* qbf   = (__bf16*)alloc((size_t)Bb * NH * Ss * HD * 2);
  __bf16* kbf   = (__bf16*)alloc((size_t)Bb * NKV * Ss * HD * 2);
  __bf16* vtb   = (__bf16*)alloc((size_t)Bb * NKV * HD * Ss * 2);
  __bf16* atbf  = (__bf16*)alloc((size_t)ROWS * (NH * HD) * 2);
  float*  x1f   = (float*) alloc((size_t)ROWS * DIMc * 4);
  __bf16* xn2bf = (__bf16*)alloc((size_t)ROWS * DIMc * 2);
  __bf16* hid   = (__bf16*)alloc((size_t)ROWS * FFNc * 2);

  const int T = 256;
  // 1) weights -> bf16 transposed [N][K]
  transpose_w_kernel<<<(DIMc * NH * HD) / T, T, 0, stream>>>(wq, wqT, DIMc, NH * HD);
  transpose_w_kernel<<<(DIMc * NKV * HD) / T, T, 0, stream>>>(wk, wkT, DIMc, NKV * HD);
  transpose_w_kernel<<<(DIMc * NKV * HD) / T, T, 0, stream>>>(wv, wvT, DIMc, NKV * HD);
  transpose_w_kernel<<<(NH * HD * DIMc) / T, T, 0, stream>>>(wo, woT, NH * HD, DIMc);
  transpose_w_kernel<<<(DIMc * FFNc) / T, T, 0, stream>>>(wg, wgT, DIMc, FFNc);
  transpose_w_kernel<<<(DIMc * FFNc) / T, T, 0, stream>>>(wu, wuT, DIMc, FFNc);
  transpose_w_kernel<<<(FFNc * DIMc) / T, T, 0, stream>>>(wd, wdT, FFNc, DIMc);
  // 2) rmsnorm1
  rmsnorm_kernel<<<ROWS, T, 0, stream>>>(x, w_norm1, xnbf, DIMc);
  // 3) q/k/v projections (WMMA)
  gemm_bf16_kernel<0><<<dim3((NH * HD) / 64, ROWS / 256), T, 0, stream>>>(
      xnbf, wqT, qf, nullptr, nullptr, ROWS, NH * HD, DIMc);
  gemm_bf16_kernel<0><<<dim3((NKV * HD) / 64, ROWS / 256), T, 0, stream>>>(
      xnbf, wkT, kf, nullptr, nullptr, ROWS, NKV * HD, DIMc);
  gemm_bf16_kernel<0><<<dim3((NKV * HD) / 64, ROWS / 256), T, 0, stream>>>(
      xnbf, wvT, vf, nullptr, nullptr, ROWS, NKV * HD, DIMc);
  // 4) RoPE + V transpose
  rope_kernel<<<(Bb * NH * Ss * 32) / T, T, 0, stream>>>(qf, cosT, sinT, qbf, NH);
  rope_kernel<<<(Bb * NKV * Ss * 32) / T, T, 0, stream>>>(kf, cosT, sinT, kbf, NKV);
  vtrans_kernel<<<(Bb * NKV * HD * Ss) / T, T, 0, stream>>>(vf, vtb);
  // 5) flash attention (WMMA): B*NH*(S/16)/4 blocks of 4 waves
  flash_attn_kernel<<<(Bb * NH * (Ss / 16)) / 4, 128, 0, stream>>>(qbf, kbf, vtb, atbf);
  // 6) wo projection + residual + attn_scale
  gemm_bf16_kernel<1><<<dim3(DIMc / 64, ROWS / 256), T, 0, stream>>>(
      atbf, woT, x1f, x, a_scale, ROWS, DIMc, NH * HD);
  // 7) rmsnorm2
  rmsnorm_kernel<<<ROWS, T, 0, stream>>>(x1f, w_norm2, xn2bf, DIMc);
  // 8) fused gate/up + silu*mul
  gemm_gateup_kernel<<<dim3(FFNc / 64, ROWS / 256), T, 0, stream>>>(
      xn2bf, wgT, wuT, hid, FFNc, DIMc);
  // 9) down projection + residual + mlp_scale -> d_out
  gemm_bf16_kernel<1><<<dim3(DIMc / 64, ROWS / 256), T, 0, stream>>>(
      hid, wdT, out, x1f, m_scale, ROWS, DIMc, FFNc);
}
